// LSTMRegressor_12266426598030
// MI455X (gfx1250) — matrix operand order repaired and checked by reading, not compiled, not run on hardware
//
#include <hip/hip_runtime.h>
#include <hip/hip_bf16.h>

// ---------------------------------------------------------------------------
// Persistent 2-layer LSTM for MI455X (gfx1250, wave32, WMMA bf16->f32).
//   B=64, T=1024, IN=64, H=512, OUT=64
// 32 workgroups x 128 threads (4 waves). WG wg owns hidden columns
// [wg*16, wg*16+16); wave = batch tile. Each wave computes the 16x16 tile of
// all 4 gates for its (batch-tile, h-tile), does the LSTM cell elementwise in
// accumulator registers, keeps c1/c2 persistent in VGPRs across all steps.
// ALL packed weights needed by a WG (221 KB) are staged into LDS once and
// re-used for all 1024 steps (CDNA5 WGP has 320 KB LDS); hot-loop B reads are
// ds_load_b128. h1/h2 double-buffered in d_ws as bf16 (global, for cross-WG
// sharing); cross-WG sync = global sense barrier, 2 per step.
// ---------------------------------------------------------------------------

typedef __attribute__((ext_vector_type(16))) __bf16 v16bf;
typedef __attribute__((ext_vector_type(8)))  float  v8f;

#define NWG 32
#define TPB 128

constexpr int Bz = 64, Tz = 1024, INz = 64, Hz = 512, OUTz = 64;
constexpr int KT1 = (INz + Hz) / 32;   // 18 k-tiles  (layer1: [x|h1], K=576)
constexpr int KT2 = (2 * Hz) / 32;     // 32 k-tiles  (layer2: [h1|h2], K=1024)
constexpr int KTo = Hz / 32;           // 16 k-tiles  (head,  K=512)
constexpr int HBUF = Bz * Hz;          // elements in one h parity buffer

// LDS layout (bytes): [L1 g0..g3][L2 g0..g3][head]
constexpr unsigned L1BLK = (unsigned)KT1 * 1024u;       // 18 KB per gate
constexpr unsigned L2OFF = 4u * L1BLK;                  // 73728
constexpr unsigned L2BLK = (unsigned)KT2 * 1024u;       // 32 KB per gate
constexpr unsigned HOFF  = L2OFF + 4u * L2BLK;          // 204800
constexpr unsigned LDS_TOTAL = HOFF + (unsigned)KTo * 1024u;  // 221184

// workspace layout (bytes)
constexpr size_t OFF_BAR = 0;
constexpr size_t OFF_WB1 = 256;
constexpr size_t SZ_WB1  = (size_t)(4 * Hz / 16) * KT1 * 512 * 2;  // 2.25 MB
constexpr size_t OFF_WB2 = OFF_WB1 + SZ_WB1;
constexpr size_t SZ_WB2  = (size_t)(4 * Hz / 16) * KT2 * 512 * 2;  // 4 MB
constexpr size_t OFF_WBO = OFF_WB2 + SZ_WB2;
constexpr size_t SZ_WBO  = (size_t)(OUTz / 16) * KTo * 512 * 2;    // 64 KB
constexpr size_t OFF_H1  = OFF_WBO + SZ_WBO;
constexpr size_t SZ_H    = (size_t)2 * HBUF * 2;                   // 128 KB
constexpr size_t OFF_H2  = OFF_H1 + SZ_H;

static __device__ __forceinline__ unsigned short f2bf_bits(float f) {
  unsigned u = __float_as_uint(f);
  u += 0x7fffu + ((u >> 16) & 1u);          // round to nearest even
  return (unsigned short)(u >> 16);
}

// fast sigmoid / tanh: v_exp_f32 + v_rcp_f32
static __device__ __forceinline__ float sigm(float v) {
  return __builtin_amdgcn_rcpf(1.0f + __expf(-v));
}
static __device__ __forceinline__ float ftanh(float v) {
  return 2.0f * __builtin_amdgcn_rcpf(1.0f + __expf(-2.0f * v)) - 1.0f;
}

// B fragment (32x16 bf16) from LDS: lane reads 32 contiguous bytes.
static __device__ __forceinline__ v16bf load_bfrag_lds(const char* lds, unsigned off) {
  union { uint4 q[2]; v16bf v; } u;
  u.q[0] = *(const uint4*)(lds + off);
  u.q[1] = *(const uint4*)(lds + off + 16);
  return u.v;
}

// A fragment (16x32 bf16) from row-major bf16 hidden-state buffer (global).
static __device__ __forceinline__ v16bf load_afrag(const char* base, unsigned off) {
  union { uint4 q[2]; v16bf v; } u;
  u.q[0] = *(const uint4*)(base + off);
  u.q[1] = *(const uint4*)(base + off + 32);
  return u.v;
}

// A fragment from the f32 input x[B,T,IN]; element off = row + t*IN + 8*hi.
static __device__ __forceinline__ v16bf load_afrag_x(const float* base,
                                                     unsigned off) {
  v16bf a;
#pragma unroll
  for (int j = 0; j < 8; ++j) a[j]     = (__bf16)base[off + j];
#pragma unroll
  for (int j = 0; j < 8; ++j) a[j + 8] = (__bf16)base[off + 16 + j];
  return a;
}

static __device__ __forceinline__ v8f wmma_bf16(v16bf a, v16bf b, v8f c) {
  return __builtin_amdgcn_wmma_f32_16x16x32_bf16(false, a, false, b,
                                                 (short)0, c, false, false);
}

// Device-wide sense barrier (32 co-resident WGs; persistent-kernel pattern).
static __device__ __forceinline__ void grid_barrier(unsigned* cnt, unsigned* gen) {
  __syncthreads();
  if (threadIdx.x == 0) {
    unsigned g = __hip_atomic_load(gen, __ATOMIC_RELAXED, __HIP_MEMORY_SCOPE_AGENT);
    __threadfence();
    unsigned old = __hip_atomic_fetch_add(cnt, 1u, __ATOMIC_ACQ_REL, __HIP_MEMORY_SCOPE_AGENT);
    if (old == NWG - 1) {
      __hip_atomic_store(cnt, 0u, __ATOMIC_RELAXED, __HIP_MEMORY_SCOPE_AGENT);
      __hip_atomic_fetch_add(gen, 1u, __ATOMIC_RELEASE, __HIP_MEMORY_SCOPE_AGENT);
    } else {
      while (__hip_atomic_load(gen, __ATOMIC_ACQUIRE, __HIP_MEMORY_SCOPE_AGENT) == g) {
        __builtin_amdgcn_s_sleep(2);
      }
    }
  }
  __syncthreads();
}

// ---------------------------------------------------------------------------
// Weight packing: dst holds B-operand tiles for gates = A @ W^T, i.e.
// B[k][n] = W[n][k], W = [src0 ; src1] concatenated along K (K0 + K1).
// 2-D grid: blockIdx.x = kt, blockIdx.y = nt, 512 threads = one tile.
//   tid = lane*16 + j;  n = nt*16 + (lane&15);  k = kt*32 + (j&7) + 8*(lane>>4) + 16*(j>>3)
// ---------------------------------------------------------------------------
__global__ void pack_weights(const float* __restrict__ src0,
                             const float* __restrict__ src1,
                             int K0, int K1,
                             unsigned short* __restrict__ dst) {
  const int kt   = blockIdx.x;
  const int nt   = blockIdx.y;
  const int KT   = gridDim.x;
  const int tid  = threadIdx.x;     // 0..511
  const int j    = tid & 15;
  const int lane = tid >> 4;
  const int n    = nt * 16 + (lane & 15);
  const int k    = kt * 32 + (j & 7) + 8 * (lane >> 4) + 16 * (j >> 3);
  const float v  = (k < K0) ? src0[(size_t)n * K0 + k]
                            : src1[(size_t)n * K1 + (k - K0)];
  dst[(size_t)(nt * KT + kt) * 512 + tid] = f2bf_bits(v);
}

__global__ void init_ws(unsigned short* __restrict__ h1,
                        unsigned short* __restrict__ h2,
                        unsigned* __restrict__ bar) {
  const size_t n = (size_t)2 * HBUF;
  for (size_t i = (size_t)blockIdx.x * blockDim.x + threadIdx.x; i < n;
       i += (size_t)gridDim.x * blockDim.x) {
    h1[i] = 0;
    h2[i] = 0;
  }
  if (blockIdx.x == 0 && threadIdx.x < 32) bar[threadIdx.x] = 0;
}

// ---------------------------------------------------------------------------
__global__ __launch_bounds__(TPB) void lstm_persistent(
    const float* __restrict__ x,
    const char* __restrict__ wb1,   // packed bf16 tiles (byte-addressed)
    const char* __restrict__ wb2,
    const char* __restrict__ wbo,
    const float* __restrict__ b_ih1, const float* __restrict__ b_hh1,
    const float* __restrict__ b_ih2, const float* __restrict__ b_hh2,
    const float* __restrict__ b_out,
    unsigned short* __restrict__ h1buf,
    unsigned short* __restrict__ h2buf,
    unsigned* __restrict__ bar,
    float* __restrict__ y) {
  extern __shared__ __align__(16) char smem[];

  const unsigned lane  = threadIdx.x & 31;
  const unsigned wave  = threadIdx.x >> 5;   // 0..3 = batch tile
  const unsigned wg    = blockIdx.x;         // 0..31 = h tile
  const unsigned m0    = wave * 16;
  const unsigned htile = wg;
  const unsigned laneL = lane & 15;
  const unsigned hi    = lane >> 4;
  const unsigned col   = htile * 16 + laneL;

  // ---- stage this WG's packed weights into LDS (once; reused 1024x) ------
  {
#pragma unroll
    for (int g = 0; g < 4; ++g) {
      const uint4* src = (const uint4*)(wb1 + (size_t)((g * 32 + htile) * KT1) * 1024);
      uint4*       d   = (uint4*)(smem + (unsigned)g * L1BLK);
      for (int i = threadIdx.x; i < KT1 * 64; i += TPB) d[i] = src[i];
    }
#pragma unroll
    for (int g = 0; g < 4; ++g) {
      const uint4* src = (const uint4*)(wb2 + (size_t)((g * 32 + htile) * KT2) * 1024);
      uint4*       d   = (uint4*)(smem + L2OFF + (unsigned)g * L2BLK);
      for (int i = threadIdx.x; i < KT2 * 64; i += TPB) d[i] = src[i];
    }
    if (wg < 16) {
      const uint4* src = (const uint4*)(wbo + (size_t)((wg >> 2) * KTo) * 1024);
      uint4*       d   = (uint4*)(smem + HOFF);
      for (int i = threadIdx.x; i < KTo * 64; i += TPB) d[i] = src[i];
    }
    __syncthreads();
  }

  // fused biases for this lane's hidden column, all 4 gates, both layers
  float bia1[4], bia2[4];
#pragma unroll
  for (int g = 0; g < 4; ++g) {
    bia1[g] = b_ih1[g * Hz + col] + b_hh1[g * Hz + col];
    bia2[g] = b_ih2[g * Hz + col] + b_hh2[g * Hz + col];
  }
  // head tile assignment: WGs 0..15, wave 0: tile (omt, ont)
  const unsigned omt  = wg & 3, ont = (wg >> 2) & 3;
  const unsigned ocol = ont * 16 + laneL;
  const float    bo   = b_out[ocol];

  // ---- hoisted per-lane BYTE offsets -------------------------------------
  const char* h1c = (const char*)h1buf;
  const char* h2c = (const char*)h2buf;
  // A-fragment row offsets into h buffers, per parity:
  const unsigned aH[2] = { 2u * ((m0 + laneL) * (unsigned)Hz + 8u * hi),
                           2u * ((unsigned)HBUF + (m0 + laneL) * (unsigned)Hz + 8u * hi) };
  // head A-fragment row offsets (batch tile omt):
  const unsigned aHo[2] = { 2u * ((omt * 16 + laneL) * (unsigned)Hz + 8u * hi),
                            2u * ((unsigned)HBUF + (omt * 16 + laneL) * (unsigned)Hz + 8u * hi) };
  // h write offsets (elements; __bf16 store), per parity; row r adds Hz:
  const unsigned wH[2] = { (m0 + 8u * hi) * (unsigned)Hz + col,
                           (unsigned)HBUF + (m0 + 8u * hi) * (unsigned)Hz + col };
  // x row base for this lane's batch row (element units):
  const unsigned xRow = (m0 + laneL) * (unsigned)(Tz * INz) + 8u * hi;
  // LDS B-fragment starting byte offsets per gate (lane term folded in):
  unsigned b1s[4], b2s[4];
#pragma unroll
  for (int g = 0; g < 4; ++g) {
    b1s[g] = (unsigned)g * L1BLK + lane * 32u;
    b2s[g] = L2OFF + (unsigned)g * L2BLK + lane * 32u;
  }
  const unsigned bos = HOFF + lane * 32u;
  // y base offset for head (element units; +t*OUTz added per step):
  const unsigned yRow = (omt * 16 + 8u * hi) * (unsigned)(Tz * OUTz) + ocol;

  v8f c1 = {0, 0, 0, 0, 0, 0, 0, 0};   // persistent cell states (f32, in VGPRs)
  v8f c2 = {0, 0, 0, 0, 0, 0, 0, 0};

  unsigned* cnt = bar;
  unsigned* gen = bar + 16;

  unsigned xT = xRow;                   // advances by INz each step

  for (int t = 0; t < Tz; ++t) {
    const unsigned pr = (unsigned)(t & 1);        // read parity
    const unsigned pw = pr ^ 1u;                  // write parity

    // ---- layer 1: gates1 = [x_t | h1] @ [W_ih1;W_hh1]^T  (K = 576) --------
    v8f acc[4];
#pragma unroll
    for (int g = 0; g < 4; ++g)
#pragma unroll
      for (int r = 0; r < 8; ++r) acc[g][r] = 0.0f;

    {
      unsigned bo0 = b1s[0], bo1 = b1s[1], bo2 = b1s[2], bo3 = b1s[3];
      // peeled: 2 k-tiles from x (f32 -> bf16 convert on the fly)
#pragma unroll
      for (int kt = 0; kt < 2; ++kt) {
        v16bf a = load_afrag_x(x, xT + (unsigned)(kt * 32));
        acc[0] = wmma_bf16(a, load_bfrag_lds(smem, bo0), acc[0]);
        acc[1] = wmma_bf16(a, load_bfrag_lds(smem, bo1), acc[1]);
        acc[2] = wmma_bf16(a, load_bfrag_lds(smem, bo2), acc[2]);
        acc[3] = wmma_bf16(a, load_bfrag_lds(smem, bo3), acc[3]);
        bo0 += 1024u; bo1 += 1024u; bo2 += 1024u; bo3 += 1024u;
      }
      // 16 k-tiles from h1 (read parity)
      unsigned ao = aH[pr];
#pragma unroll 2
      for (int kt = 0; kt < 16; ++kt) {
        v16bf a = load_afrag(h1c, ao);
        acc[0] = wmma_bf16(a, load_bfrag_lds(smem, bo0), acc[0]);
        acc[1] = wmma_bf16(a, load_bfrag_lds(smem, bo1), acc[1]);
        acc[2] = wmma_bf16(a, load_bfrag_lds(smem, bo2), acc[2]);
        acc[3] = wmma_bf16(a, load_bfrag_lds(smem, bo3), acc[3]);
        ao += 64u;
        bo0 += 1024u; bo1 += 1024u; bo2 += 1024u; bo3 += 1024u;
      }
    }

    // LSTM cell 1 elementwise, entirely in accumulator registers
    {
      unsigned w = wH[pw];
#pragma unroll
      for (int r = 0; r < 8; ++r) {
        float iv = sigm(acc[0][r] + bia1[0]);
        float fv = sigm(acc[1][r] + bia1[1]);
        float gv = ftanh(acc[2][r] + bia1[2]);
        float ov = sigm(acc[3][r] + bia1[3]);
        float c  = fv * c1[r] + iv * gv;
        c1[r]    = c;
        ((__bf16*)h1buf)[w] = (__bf16)(ov * ftanh(c));
        w += (unsigned)Hz;
      }
    }

    grid_barrier(cnt, gen);   // h1(t) visible to all WGs

    // ---- layer 2: gates2 = [h1 | h2] @ [W_ih2;W_hh2]^T  (K = 1024) --------
#pragma unroll
    for (int g = 0; g < 4; ++g)
#pragma unroll
      for (int r = 0; r < 8; ++r) acc[g][r] = 0.0f;

    {
      unsigned bo0 = b2s[0], bo1 = b2s[1], bo2 = b2s[2], bo3 = b2s[3];
      // 16 k-tiles from h1 (freshly written, write parity)
      unsigned ao = aH[pw];
#pragma unroll 2
      for (int kt = 0; kt < 16; ++kt) {
        v16bf a = load_afrag(h1c, ao);
        acc[0] = wmma_bf16(a, load_bfrag_lds(smem, bo0), acc[0]);
        acc[1] = wmma_bf16(a, load_bfrag_lds(smem, bo1), acc[1]);
        acc[2] = wmma_bf16(a, load_bfrag_lds(smem, bo2), acc[2]);
        acc[3] = wmma_bf16(a, load_bfrag_lds(smem, bo3), acc[3]);
        ao += 64u;
        bo0 += 1024u; bo1 += 1024u; bo2 += 1024u; bo3 += 1024u;
      }
      // 16 k-tiles from h2 (read parity)
      ao = aH[pr];
#pragma unroll 2
      for (int kt = 0; kt < 16; ++kt) {
        v16bf a = load_afrag(h2c, ao);
        acc[0] = wmma_bf16(a, load_bfrag_lds(smem, bo0), acc[0]);
        acc[1] = wmma_bf16(a, load_bfrag_lds(smem, bo1), acc[1]);
        acc[2] = wmma_bf16(a, load_bfrag_lds(smem, bo2), acc[2]);
        acc[3] = wmma_bf16(a, load_bfrag_lds(smem, bo3), acc[3]);
        ao += 64u;
        bo0 += 1024u; bo1 += 1024u; bo2 += 1024u; bo3 += 1024u;
      }
    }

    {
      unsigned w = wH[pw];
#pragma unroll
      for (int r = 0; r < 8; ++r) {
        float iv = sigm(acc[0][r] + bia2[0]);
        float fv = sigm(acc[1][r] + bia2[1]);
        float gv = ftanh(acc[2][r] + bia2[2]);
        float ov = sigm(acc[3][r] + bia2[3]);
        float c  = fv * c2[r] + iv * gv;
        c2[r]    = c;
        ((__bf16*)h2buf)[w] = (__bf16)(ov * ftanh(c));
        w += (unsigned)Hz;
      }
    }

    grid_barrier(cnt, gen);   // h2(t) visible to all WGs

    // ---- head: y_t = h2 @ W_out^T + b_out  (16 tiles; WGs 0..15, wave 0) --
    if (wg < 16 && wave == 0) {
      v8f oa = {0, 0, 0, 0, 0, 0, 0, 0};
      unsigned bb = bos;
      unsigned ao = aHo[pw];  // new h2
#pragma unroll 2
      for (int kt = 0; kt < KTo; ++kt) {
        v16bf a = load_afrag(h2c, ao);
        oa = wmma_bf16(a, load_bfrag_lds(smem, bb), oa);
        ao += 64u;
        bb += 1024u;
      }
      unsigned yo = yRow + (unsigned)(t * OUTz);
#pragma unroll
      for (int r = 0; r < 8; ++r) {
        y[yo] = oa[r] + bo;
        yo += (unsigned)(Tz * OUTz);
      }
    }
    xT += (unsigned)INz;
    // next barrier (step t+1, barrier 1) also fences the head vs h2 rewrite
  }
}

// ---------------------------------------------------------------------------
extern "C" void kernel_launch(void* const* d_in, const int* in_sizes, int n_in,
                              void* d_out, int out_size, void* d_ws, size_t ws_size,
                              hipStream_t stream) {
  const float* x     = (const float*)d_in[0];
  const float* W_ih1 = (const float*)d_in[1];
  const float* W_hh1 = (const float*)d_in[2];
  const float* b_ih1 = (const float*)d_in[3];
  const float* b_hh1 = (const float*)d_in[4];
  const float* W_ih2 = (const float*)d_in[5];
  const float* W_hh2 = (const float*)d_in[6];
  const float* b_ih2 = (const float*)d_in[7];
  const float* b_hh2 = (const float*)d_in[8];
  const float* W_out = (const float*)d_in[9];
  const float* b_out = (const float*)d_in[10];
  float* y = (float*)d_out;

  char* ws = (char*)d_ws;
  unsigned*       bar = (unsigned*)(ws + OFF_BAR);
  unsigned short* wb1 = (unsigned short*)(ws + OFF_WB1);
  unsigned short* wb2 = (unsigned short*)(ws + OFF_WB2);
  unsigned short* wbo = (unsigned short*)(ws + OFF_WBO);
  unsigned short* h1  = (unsigned short*)(ws + OFF_H1);
  unsigned short* h2  = (unsigned short*)(ws + OFF_H2);

  init_ws<<<64, 256, 0, stream>>>(h1, h2, bar);
  pack_weights<<<dim3(KT1, 4 * Hz / 16), 512, 0, stream>>>(W_ih1, W_hh1, INz, Hz, wb1);
  pack_weights<<<dim3(KT2, 4 * Hz / 16), 512, 0, stream>>>(W_ih2, W_hh2, Hz, Hz, wb2);
  pack_weights<<<dim3(KTo, OUTz / 16), 512, 0, stream>>>(W_out, W_out, Hz, 0, wbo);

  lstm_persistent<<<NWG, TPB, LDS_TOTAL, stream>>>(
      x, (const char*)wb1, (const char*)wb2, (const char*)wbo,
      b_ih1, b_hh1, b_ih2, b_hh2, b_out, h1, h2, bar, y);
}
